// convex_monotone_network_75711683494298
// MI455X (gfx1250) — matrix operand order repaired and checked by reading, not compiled, not run on hardware
//
#include <hip/hip_runtime.h>
#include <hip/hip_bf16.h>
#include <math.h>

typedef _Float16 f16;
typedef __attribute__((ext_vector_type(16))) _Float16 v16h;
typedef __attribute__((ext_vector_type(8)))  float    v8f;

union FragH { v16h v; unsigned int u[8]; };
union AccF  { v8f  v; float f[8]; };

__device__ __forceinline__ v8f wmma_f16(v16h a, v16h b, v8f c) {
  // 8 args: (neg_a, A, neg_b, B, c_mod, C, reuse_a, reuse_b)
  return __builtin_amdgcn_wmma_f32_16x16x32_f16(false, a, false, b, (short)0, c,
                                                false, false);
}

// Fast activation pair: e = exp(-|a|) shared between sigmoid and softplus.
//   sigmoid(a)  = rcp(1+e)            (a >= 0)
//               = e * rcp(1+e)        (a <  0)
//   softplus(a) = max(a,0) + log(1+e)
__device__ __forceinline__ void act_pair(float a, float& sig, float& sp) {
  float e = __expf(-fabsf(a));
  float r = __builtin_amdgcn_rcpf(1.0f + e);
  sig = (a >= 0.0f) ? r : e * r;
  sp  = fmaxf(a, 0.0f) + __logf(1.0f + e);
}

__device__ __forceinline__ float fast_sigmoid(float a) {
  float e = __expf(-fabsf(a));
  float r = __builtin_amdgcn_rcpf(1.0f + e);
  return (a >= 0.0f) ? r : e * r;
}

// ---- Fragment loaders (ISA 7.12.2 16-bit layouts, wave32) -------------------
// A: 16x32 (MxK). lane&15 = M row; lanes>=16 add +8 to K within each 16-block.
// VGPR v holds packed pair (K0, K0+1), K0 = (v&3)*2 + (v>>2)*16.
__device__ __forceinline__ v16h load_fragA(const f16* base, int ld, int mbase,
                                           int kbase, int lane) {
  FragH f;
  const f16* rp = base + (mbase + (lane & 15)) * ld + kbase + ((lane >> 4) << 3);
#pragma unroll
  for (int v = 0; v < 8; ++v) {
    int k0 = ((v & 3) << 1) + ((v >> 2) << 4);
    f.u[v] = *(const unsigned int*)(rp + k0);
  }
  return f.v;
}

// B: 32x16 (KxN), source stored [N][K] row-major with leading dim ld.
// lane&15 = N col; lanes>=16 add +8 to K within each 16-block; same VGPR pairs as A.
__device__ __forceinline__ v16h load_fragB(const f16* base, int ld, int nbase,
                                           int kbase, int lane) {
  FragH f;
  const f16* cp = base + (nbase + (lane & 15)) * ld + kbase + ((lane >> 4) << 3);
#pragma unroll
  for (int v = 0; v < 8; ++v) f.u[v] = *(const unsigned int*)(cp + ((v & 3) << 1) + ((v >> 2) << 4));
  return f.v;
}

__device__ __forceinline__ void zero_acc(v8f acc[2][4]) {
  v8f z = {};
#pragma unroll
  for (int mt = 0; mt < 2; ++mt)
#pragma unroll
    for (int nt = 0; nt < 4; ++nt) acc[mt][nt] = z;
}

// C += A[32xK] * B[KxN-slab]; wave covers columns [nwave, nwave+64).
template <int KTOT>
__device__ __forceinline__ void gemm8(v8f acc[2][4], const f16* A, int lda,
                                      const f16* W, int nwave, int lane) {
  for (int k = 0; k < KTOT; k += 32) {
    v16h a0 = load_fragA(A, lda, 0, k, lane);
    v16h a1 = load_fragA(A, lda, 16, k, lane);
#pragma unroll
    for (int nt = 0; nt < 4; ++nt) {
      v16h b = load_fragB(W, KTOT, nwave + nt * 16, k, lane);
      acc[0][nt] = wmma_f16(a0, b, acc[0][nt]);
      acc[1][nt] = wmma_f16(a1, b, acc[1][nt]);
    }
  }
}

// Forward epilogue: a = acc + bias; store sigmoid(a) (always) and softplus(a).
__device__ __forceinline__ void fwd_epilogue(v8f acc[2][4], const float* bias,
                                             int nwave, int lane, f16* sdst,
                                             f16* zdst) {
#pragma unroll
  for (int mt = 0; mt < 2; ++mt)
#pragma unroll
    for (int nt = 0; nt < 4; ++nt) {
      AccF c; c.v = acc[mt][nt];
      int col = nwave + nt * 16 + (lane & 15);
      float bv = bias[col];
#pragma unroll
      for (int e = 0; e < 8; ++e) {
        int m = mt * 16 + e + ((lane >> 4) << 3);
        float a = c.f[e] + bv;
        if (zdst) {
          float s, z;
          act_pair(a, s, z);
          sdst[m * 512 + col] = (f16)s;
          zdst[m * 512 + col] = (f16)z;
        } else {
          sdst[m * 512 + col] = (f16)fast_sigmoid(a);
        }
      }
    }
}

// Backward epilogue: g = acc * sigmoid (elementwise), store f16.
__device__ __forceinline__ void bwd_epilogue(v8f acc[2][4], const f16* s,
                                             int nwave, int lane, f16* gdst) {
#pragma unroll
  for (int mt = 0; mt < 2; ++mt)
#pragma unroll
    for (int nt = 0; nt < 4; ++nt) {
      AccF c; c.v = acc[mt][nt];
      int col = nwave + nt * 16 + (lane & 15);
#pragma unroll
      for (int e = 0; e < 8; ++e) {
        int m = mt * 16 + e + ((lane >> 4) << 3);
        gdst[m * 512 + col] = (f16)(c.f[e] * (float)s[m * 512 + col]);
      }
    }
}

// dx tile (16x16, N=64 split over 8 waves): acc += G[32x512] * W[512x64 as NK].
__device__ __forceinline__ void dx_gemm(v8f& acc, const f16* G, const f16* W,
                                        int mt, int nt, int lane) {
  for (int k = 0; k < 512; k += 32) {
    v16h a = load_fragA(G, 512, mt * 16, k, lane);
    v16h b = load_fragB(W, 512, nt * 16, k, lane);
    acc = wmma_f16(a, b, acc);
  }
}

// ---- Fused ICNN forward + input-gradient kernel -----------------------------
__global__ __launch_bounds__(256) void icnn_grad_kernel(
    const float* __restrict__ state, const float* __restrict__ bz0,
    const float* __restrict__ bx0, const float* __restrict__ bx1,
    const float* __restrict__ bx2, const float* __restrict__ WzL,
    const float* __restrict__ WxL, const f16* __restrict__ Wz0f,
    const f16* __restrict__ Wx0f, const f16* __restrict__ Wx1f,
    const f16* __restrict__ Wx2f, const f16* __restrict__ Wz1f,
    const f16* __restrict__ Wz2f, const f16* __restrict__ Wz3f,
    const f16* __restrict__ Wz1T, const f16* __restrict__ Wz2T,
    const f16* __restrict__ Wz3T, const f16* __restrict__ Wz0T,
    const f16* __restrict__ Wx0T, const f16* __restrict__ Wx1T,
    const f16* __restrict__ Wx2T, float* __restrict__ out, int Btot) {
  extern __shared__ f16 smem[];
  f16* xl = smem;            // [32][64]
  f16* zA = smem + 2048;     // [32][512] ping
  f16* zB = zA + 16384;      // [32][512] pong
  f16* s1 = zB + 16384;      // sigmoids per layer, [32][512] each
  f16* s2 = s1 + 16384;
  f16* s3 = s2 + 16384;
  f16* s4 = s3 + 16384;

  const int tid = threadIdx.x;
  const int lane = tid & 31;
  const int wave = tid >> 5;
  const int nwave = wave * 64;
  const long row0 = (long)blockIdx.x * 32;

  // x = state - 1 -> LDS f16
  for (int i = tid; i < 32 * 64; i += 256) {
    long g = row0 * 64 + i;
    xl[i] = (f16)((g < (long)Btot * 64 ? state[g] : 0.0f) - 1.0f);
  }
  __syncthreads();

  v8f acc[2][4];

  // L1: a1 = x @ Wz0^T + bz0
  zero_acc(acc);
  gemm8<64>(acc, xl, 64, Wz0f, nwave, lane);
  fwd_epilogue(acc, bz0, nwave, lane, s1, zA);
  __syncthreads();

  // L2: a2 = z1 @ Wz1^T + x @ Wx0^T + bx0
  zero_acc(acc);
  gemm8<512>(acc, zA, 512, Wz1f, nwave, lane);
  gemm8<64>(acc, xl, 64, Wx0f, nwave, lane);
  fwd_epilogue(acc, bx0, nwave, lane, s2, zB);
  __syncthreads();

  // L3
  zero_acc(acc);
  gemm8<512>(acc, zB, 512, Wz2f, nwave, lane);
  gemm8<64>(acc, xl, 64, Wx1f, nwave, lane);
  fwd_epilogue(acc, bx1, nwave, lane, s3, zA);
  __syncthreads();

  // L4: only sigmoid needed (grad path); z4 itself is dead.
  zero_acc(acc);
  gemm8<512>(acc, zA, 512, Wz3f, nwave, lane);
  gemm8<64>(acc, xl, 64, Wx2f, nwave, lane);
  fwd_epilogue(acc, bx2, nwave, lane, s4, (f16*)nullptr);
  __syncthreads();

  // g4 = WzL * s4 -> zB
  for (int i = tid; i < 32 * 512; i += 256)
    zB[i] = (f16)(WzL[i & 511] * (float)s4[i]);
  __syncthreads();

  const int mt_dx = wave >> 2, nt_dx = wave & 3;
  v8f dxa = {};
  dx_gemm(dxa, zB, Wx2T, mt_dx, nt_dx, lane);   // dx += g4 @ Wx2

  // g3 = (g4 @ Wz3) * s3 -> zA
  zero_acc(acc);
  gemm8<512>(acc, zB, 512, Wz3T, nwave, lane);
  bwd_epilogue(acc, s3, nwave, lane, zA);
  __syncthreads();
  dx_gemm(dxa, zA, Wx1T, mt_dx, nt_dx, lane);   // dx += g3 @ Wx1

  // g2 -> zB
  zero_acc(acc);
  gemm8<512>(acc, zA, 512, Wz2T, nwave, lane);
  bwd_epilogue(acc, s2, nwave, lane, zB);
  __syncthreads();
  dx_gemm(dxa, zB, Wx0T, mt_dx, nt_dx, lane);   // dx += g2 @ Wx0

  // g1 -> zA
  zero_acc(acc);
  gemm8<512>(acc, zB, 512, Wz1T, nwave, lane);
  bwd_epilogue(acc, s1, nwave, lane, zA);
  __syncthreads();
  dx_gemm(dxa, zA, Wz0T, mt_dx, nt_dx, lane);   // dx += g1 @ Wz0

  // out = dx + WxL
  AccF r; r.v = dxa;
  int col = nt_dx * 16 + (lane & 15);
  float wl = WxL[col];
#pragma unroll
  for (int e = 0; e < 8; ++e) {
    int m = mt_dx * 16 + e + ((lane >> 4) << 3);
    if (row0 + m < Btot) out[(row0 + m) * 64 + col] = r.f[e] + wl;
  }
}

// ---- Weight precision/layout conversion -------------------------------------
__global__ void cvt_kernel(const float* __restrict__ src, f16* __restrict__ dst,
                           int rows, int cols, int transpose) {
  int i = blockIdx.x * blockDim.x + threadIdx.x;
  int n = rows * cols;
  if (i >= n) return;
  float v = src[i];
  if (transpose) {
    int r = i / cols, c = i - r * cols;
    dst[c * rows + r] = (f16)v;
  } else {
    dst[i] = (f16)v;
  }
}

extern "C" void kernel_launch(void* const* d_in, const int* in_sizes, int n_in,
                              void* d_out, int out_size, void* d_ws,
                              size_t ws_size, hipStream_t stream) {
  const float* state = (const float*)d_in[0];
  const float* Wz0 = (const float*)d_in[1];
  const float* bz0 = (const float*)d_in[2];
  const float* Wz1 = (const float*)d_in[3];
  const float* Wz2 = (const float*)d_in[4];
  const float* Wz3 = (const float*)d_in[5];
  const float* WzL = (const float*)d_in[6];
  const float* Wx0 = (const float*)d_in[7];
  const float* bx0 = (const float*)d_in[8];
  const float* Wx1 = (const float*)d_in[9];
  const float* bx1 = (const float*)d_in[10];
  const float* Wx2 = (const float*)d_in[11];
  const float* bx2 = (const float*)d_in[12];
  const float* WxL = (const float*)d_in[13];
  float* out = (float*)d_out;

  // Workspace layout (f16 elements)
  f16* ws = (f16*)d_ws;
  f16* Wz0f = ws;                 // [512][64]
  f16* Wx0f = Wz0f + 32768;       // [512][64]
  f16* Wx1f = Wx0f + 32768;
  f16* Wx2f = Wx1f + 32768;
  f16* Wz1f = Wx2f + 32768;       // [512][512]
  f16* Wz2f = Wz1f + 262144;
  f16* Wz3f = Wz2f + 262144;
  f16* Wz1T = Wz3f + 262144;      // transposed [512][512]
  f16* Wz2T = Wz1T + 262144;
  f16* Wz3T = Wz2T + 262144;
  f16* Wz0T = Wz3T + 262144;      // [64][512]
  f16* Wx0T = Wz0T + 32768;
  f16* Wx1T = Wx0T + 32768;
  f16* Wx2T = Wx1T + 32768;

  auto cvt = [&](const float* s, f16* d, int r, int c, int t) {
    int n = r * c;
    cvt_kernel<<<(n + 255) / 256, 256, 0, stream>>>(s, d, r, c, t);
  };
  cvt(Wz0, Wz0f, 512, 64, 0);
  cvt(Wx0, Wx0f, 512, 64, 0);
  cvt(Wx1, Wx1f, 512, 64, 0);
  cvt(Wx2, Wx2f, 512, 64, 0);
  cvt(Wz1, Wz1f, 512, 512, 0);
  cvt(Wz2, Wz2f, 512, 512, 0);
  cvt(Wz3, Wz3f, 512, 512, 0);
  cvt(Wz1, Wz1T, 512, 512, 1);
  cvt(Wz2, Wz2T, 512, 512, 1);
  cvt(Wz3, Wz3T, 512, 512, 1);
  cvt(Wz0, Wz0T, 512, 64, 1);
  cvt(Wx0, Wx0T, 512, 64, 1);
  cvt(Wx1, Wx1T, 512, 64, 1);
  cvt(Wx2, Wx2T, 512, 64, 1);

  const int Btot = in_sizes[0] / 64;
  const int nblk = (Btot + 31) / 32;
  const size_t SMEM = (size_t)(2048 + 6 * 16384) * sizeof(f16);  // 200704 B
  (void)hipFuncSetAttribute((const void*)icnn_grad_kernel,
                            hipFuncAttributeMaxDynamicSharedMemorySize,
                            (int)SMEM);
  icnn_grad_kernel<<<nblk, 256, SMEM, stream>>>(
      state, bz0, bx0, bx1, bx2, WzL, WxL, Wz0f, Wx0f, Wx1f, Wx2f, Wz1f, Wz2f,
      Wz3f, Wz1T, Wz2T, Wz3T, Wz0T, Wx0T, Wx1T, Wx2T, out, Btot);
}
